// TransformerBlock_5162550689936
// MI455X (gfx1250) — compile-verified, matrix operand order checked by reading, not run on hardware
//
#include <hip/hip_runtime.h>
#include <hip/hip_bf16.h>
#include <math.h>

// ---------------------------------------------------------------------------
// MI455X (gfx1250) transformer block.
// All GEMMs run on v_wmma_f32_16x16x32_bf16 (wave32, 16x16x32 tiles).
// Tile staging uses GLOBAL_LOAD_ASYNC_TO_LDS_B128 (ASYNCcnt) with double
// buffering when the toolchain exposes the gfx1250 async builtins.
// ---------------------------------------------------------------------------

typedef __bf16 bf16;
typedef bf16  bf16x8 __attribute__((ext_vector_type(8)));
typedef bf16  v16bf  __attribute__((ext_vector_type(16)));
typedef float v8f    __attribute__((ext_vector_type(8)));

#define HW_PIX   16384              // 128*128
#define NBATCH   8
#define MTOK     (NBATCH * HW_PIX)  // 131072 tokens
#define CDIM     128
#define HIDDEN   340
#define HIDP     352                // HIDDEN padded to multiple of 32
#define LSTR     40                 // LDS row stride in bf16 (80B, 16B aligned)

#if __has_builtin(__builtin_amdgcn_global_load_async_to_lds_b128) && \
    __has_builtin(__builtin_amdgcn_s_wait_asynccnt)
#define HAVE_ASYNC 1
#else
#define HAVE_ASYNC 0
#endif

#if HAVE_ASYNC
// Builtin prototype (from hipcc diagnostic): non-const pointer to a 16-byte
// int vector, AS1 (__device__) source / AS3 (__shared__) LDS destination.
typedef int i32x4g __attribute__((vector_size(16)));
#define AS_GLOBAL __attribute__((address_space(1)))
#define AS_LDS    __attribute__((address_space(3)))
__device__ inline void async_copy16(const bf16* g, bf16* l) {
  __builtin_amdgcn_global_load_async_to_lds_b128(
      (AS_GLOBAL i32x4g*)g, (AS_LDS i32x4g*)l, 0, 0);
}
#endif

__device__ inline v16bf make_frag(bf16x8 lo, bf16x8 hi) {
  v16bf r;
#pragma unroll
  for (int i = 0; i < 8; ++i) { r[i] = lo[i]; r[i + 8] = hi[i]; }
  return r;
}

// Fragment gather + 2x4 WMMA tile block for one K-step of 32.
// 16-bit A 16x32 layout: lane = (row, K-half); elements 0..7 hold
// K = fh*8..+7, elements 8..15 hold K = fh*8+16..+7  ->  two b128 LDS reads.
__device__ inline void wmma_step(const bf16* At, const bf16* Bt,
                                 int wm, int wn, int fr, int fh,
                                 v8f acc[2][4]) {
  v16bf afrag[2], bfrag[4];
#pragma unroll
  for (int mi = 0; mi < 2; ++mi) {
    const bf16* p = &At[(wm + mi * 16 + fr) * LSTR + fh * 8];
    afrag[mi] = make_frag(*(const bf16x8*)p, *(const bf16x8*)(p + 16));
  }
#pragma unroll
  for (int ni = 0; ni < 4; ++ni) {
    const bf16* p = &Bt[(wn + ni * 16 + fr) * LSTR + fh * 8];
    bfrag[ni] = make_frag(*(const bf16x8*)p, *(const bf16x8*)(p + 16));
  }
#pragma unroll
  for (int mi = 0; mi < 2; ++mi)
#pragma unroll
    for (int ni = 0; ni < 4; ++ni)
      acc[mi][ni] = __builtin_amdgcn_wmma_f32_16x16x32_bf16(
          false, afrag[mi], false, bfrag[ni], (short)0, acc[mi][ni], false, false);
}

// ---------------------------------------------------------------------------
// Generic bf16 WMMA GEMM:  C[M,N] = A[M,K] * W[N,K]^T  (+bias) (+residual)
// Block = 256 threads = 8 waves; block tile 128x128; K step 32.
// Wave w -> rows [(w&3)*32, +32), cols [(w>>2)*64, +64): 2x4 WMMA tiles.
// MODE 0: bf16 output, row stride ldc.
// MODE 1: f32 output in BCHW layout (N==128 channels), += residual (BCHW f32).
// ---------------------------------------------------------------------------
template <int MODE>
__global__ __launch_bounds__(256) void gemm_wmma(
    const bf16*  __restrict__ A,  const bf16* __restrict__ Bw,
    const float* __restrict__ bias, const float* __restrict__ resid,
    bf16* __restrict__ outB, float* __restrict__ outF,
    int M, int N, int K, int ldc)
{
  const int tid  = threadIdx.x;
  const int m0   = blockIdx.y << 7;
  const int n0   = blockIdx.x << 7;
  const int wave = tid >> 5;
  const int lane = tid & 31;
  const int wm   = (wave & 3) << 5;
  const int wn   = (wave >> 2) << 6;
  const int fr   = lane & 15;   // fragment row (A) / col (B)
  const int fh   = lane >> 4;   // K-half selector

  v8f acc[2][4];
#pragma unroll
  for (int i = 0; i < 2; ++i)
#pragma unroll
    for (int j = 0; j < 4; ++j)
#pragma unroll
      for (int e = 0; e < 8; ++e) acc[i][j][e] = 0.f;

  // cooperative tile staging: 2 threads per row, 16 bf16 (two 16B chunks) each
  const int lrow  = tid >> 1;
  const int lhalf = (tid & 1) << 4;
  const int loff  = lrow * LSTR + lhalf;

#if HAVE_ASYNC
  // Double-buffered async staging. Clamp out-of-range rows to a valid source:
  // garbage only lands in columns >= N which the epilogue masks.
  __shared__ bf16 At[2][128 * LSTR];
  __shared__ bf16 Bt[2][128 * LSTR];
  const int arow = (m0 + lrow < M) ? (m0 + lrow) : (M - 1);
  const int brow = (n0 + lrow < N) ? (n0 + lrow) : (N - 1);
  const bf16* aptr = A  + (size_t)arow * K + lhalf;
  const bf16* bptr = Bw + (size_t)brow * K + lhalf;
  const int nk = K >> 5;

  async_copy16(aptr,     &At[0][loff]);
  async_copy16(aptr + 8, &At[0][loff + 8]);
  async_copy16(bptr,     &Bt[0][loff]);
  async_copy16(bptr + 8, &Bt[0][loff + 8]);

  for (int it = 0; it < nk; ++it) {
    const int cur = it & 1;
    if (it + 1 < nk) {
      const int kn = (it + 1) << 5;
      async_copy16(aptr + kn,     &At[cur ^ 1][loff]);
      async_copy16(aptr + kn + 8, &At[cur ^ 1][loff + 8]);
      async_copy16(bptr + kn,     &Bt[cur ^ 1][loff]);
      async_copy16(bptr + kn + 8, &Bt[cur ^ 1][loff + 8]);
      __builtin_amdgcn_s_wait_asynccnt(4);   // oldest tile (4 async ops) done
    } else {
      __builtin_amdgcn_s_wait_asynccnt(0);
    }
    __syncthreads();
    wmma_step(At[cur], Bt[cur], wm, wn, fr, fh, acc);
    __syncthreads();                          // protect buffer before reuse
  }
#else
  // Fallback: load -> VGPR -> ds_store staging.
  __shared__ bf16 At[1][128 * LSTR];
  __shared__ bf16 Bt[1][128 * LSTR];
  const bf16* aptr = A  + (size_t)(m0 + lrow) * K + lhalf;
  const bf16* bptr = Bw + (size_t)(n0 + lrow) * K + lhalf;
  const bool aval  = (m0 + lrow) < M;
  const bool bval  = (n0 + lrow) < N;
  for (int k0 = 0; k0 < K; k0 += 32) {
    uint4 a0 = {0,0,0,0}, a1 = {0,0,0,0}, b0 = {0,0,0,0}, b1 = {0,0,0,0};
    if (aval) { a0 = *(const uint4*)(aptr + k0); a1 = *(const uint4*)(aptr + k0 + 8); }
    if (bval) { b0 = *(const uint4*)(bptr + k0); b1 = *(const uint4*)(bptr + k0 + 8); }
    if (k0 + 32 < K) {
      if (aval) __builtin_prefetch(aptr + k0 + 32, 0, 1);
      if (bval) __builtin_prefetch(bptr + k0 + 32, 0, 1);
    }
    *(uint4*)&At[0][loff]     = a0;
    *(uint4*)&At[0][loff + 8] = a1;
    *(uint4*)&Bt[0][loff]     = b0;
    *(uint4*)&Bt[0][loff + 8] = b1;
    __syncthreads();
    wmma_step(At[0], Bt[0], wm, wn, fr, fh, acc);
    __syncthreads();
  }
#endif

  // Epilogue. C/D layout: VGPR j holds row j + 8*fh, col fr within the tile.
#pragma unroll
  for (int mi = 0; mi < 2; ++mi)
#pragma unroll
    for (int ni = 0; ni < 4; ++ni)
#pragma unroll
      for (int j = 0; j < 8; ++j) {
        int row = m0 + wm + mi * 16 + (fh << 3) + j;
        int col = n0 + wn + ni * 16 + fr;
        if (row < M && col < N) {
          float v = acc[mi][ni][j];
          if (bias) v += bias[col];
          if (MODE == 0) {
            outB[(size_t)row * ldc + col] = (bf16)v;
          } else {
            // token row = b*16384 + pix; channel = col; BCHW f32
            size_t off = ((size_t)(row >> 14) << 21) + ((size_t)col << 14)
                       + (size_t)(row & (HW_PIX - 1));
            outF[off] = v + resid[off];
          }
        }
      }
}

// ---------------------------------------------------------------------------
// LayerNorm over channels (BCHW f32 in) -> token-major bf16 [M, 128] out.
// ---------------------------------------------------------------------------
__global__ __launch_bounds__(256) void ln_kernel(
    const float* __restrict__ x, const float* __restrict__ w,
    const float* __restrict__ b, bf16* __restrict__ out)
{
  int t   = blockIdx.x * 256 + threadIdx.x;
  int bi  = t >> 14;
  int pix = t & (HW_PIX - 1);
  const float* xp = x + ((size_t)bi << 21) + pix;
  float s = 0.f, s2 = 0.f;
#pragma unroll 8
  for (int c = 0; c < CDIM; ++c) {
    float v = xp[(size_t)c << 14];
    s += v; s2 += v * v;
  }
  float mu  = s * (1.f / 128.f);
  float var = s2 * (1.f / 128.f) - mu * mu;
  float r   = rsqrtf(var + 1e-5f);
  bf16* op = out + (size_t)t * CDIM;
#pragma unroll 8
  for (int c = 0; c < CDIM; ++c) {
    float v = xp[(size_t)c << 14];
    op[c] = (bf16)((v - mu) * r * w[c] + b[c]);
  }
}

// ---------------------------------------------------------------------------
// Attention pass A: per (b, head) reduce kv[8][8], ksum[8], vsum[8] over L.
// qkvo layout: [M, 512] bf16; q cols 0.., k cols 128.., v cols 256.., o 384..
// ---------------------------------------------------------------------------
__global__ __launch_bounds__(256) void attn_reduce_kernel(
    const bf16* __restrict__ qkvo, float* __restrict__ stats)
{
  const int b  = blockIdx.x >> 4;
  const int nh = blockIdx.x & 15;
  float kv[8][8], ksum[8], vsum[8];
#pragma unroll
  for (int d = 0; d < 8; ++d) {
    ksum[d] = 0.f; vsum[d] = 0.f;
#pragma unroll
    for (int e = 0; e < 8; ++e) kv[d][e] = 0.f;
  }
  for (int l = threadIdx.x; l < HW_PIX; l += 256) {
    size_t m = (size_t)b * HW_PIX + l;
    const bf16* base = qkvo + (m << 9);
    bf16x8 kraw = *(const bf16x8*)(base + 128 + nh * 8);
    bf16x8 vraw = *(const bf16x8*)(base + 256 + nh * 8);
    float kf[8], vv[8];
#pragma unroll
    for (int d = 0; d < 8; ++d) {
      float xk = (float)kraw[d];
      kf[d] = xk > 0.f ? xk + 1.f : __expf(xk);   // elu(x)+1
      vv[d] = (float)vraw[d];
      ksum[d] += kf[d];
      vsum[d] += vv[d];
    }
    // RoPE: angles {1,1,1e-4,1e-4}; dims 0..3 use h, dims 4..7 use w
    int h = l >> 7, wq = l & 127;
    float fhh = (float)h, fww = (float)wq;
    float sn[8], cs[8];
    sn[0] = sn[1] = __sinf(fhh);         cs[0] = cs[1] = __cosf(fhh);
    sn[2] = sn[3] = __sinf(fhh * 1e-4f); cs[2] = cs[3] = __cosf(fhh * 1e-4f);
    sn[4] = sn[5] = __sinf(fww);         cs[4] = cs[5] = __cosf(fww);
    sn[6] = sn[7] = __sinf(fww * 1e-4f); cs[6] = cs[7] = __cosf(fww * 1e-4f);
    float kr[8];
#pragma unroll
    for (int pr = 0; pr < 4; ++pr) {
      float x0 = kf[2 * pr], x1 = kf[2 * pr + 1];
      kr[2 * pr]     = x0 * cs[2 * pr]     - x1 * sn[2 * pr];
      kr[2 * pr + 1] = x1 * cs[2 * pr + 1] + x0 * sn[2 * pr + 1];
    }
#pragma unroll
    for (int d = 0; d < 8; ++d)
#pragma unroll
      for (int e = 0; e < 8; ++e) kv[d][e] += kr[d] * vv[e];
  }
  // wave32 shuffle reduction, then cross-wave LDS atomics
#pragma unroll
  for (int d = 0; d < 8; ++d) {
#pragma unroll
    for (int e = 0; e < 8; ++e)
      for (int off = 16; off > 0; off >>= 1)
        kv[d][e] += __shfl_xor(kv[d][e], off, 32);
    for (int off = 16; off > 0; off >>= 1) {
      ksum[d] += __shfl_xor(ksum[d], off, 32);
      vsum[d] += __shfl_xor(vsum[d], off, 32);
    }
  }
  __shared__ float red[80];
  if (threadIdx.x < 80) red[threadIdx.x] = 0.f;
  __syncthreads();
  if ((threadIdx.x & 31) == 0) {
#pragma unroll
    for (int d = 0; d < 8; ++d) {
#pragma unroll
      for (int e = 0; e < 8; ++e) atomicAdd(&red[d * 8 + e], kv[d][e]);
      atomicAdd(&red[64 + d], ksum[d]);
      atomicAdd(&red[72 + d], vsum[d]);
    }
  }
  __syncthreads();
  if (threadIdx.x < 80)
    stats[(size_t)blockIdx.x * 80 + threadIdx.x] = red[threadIdx.x];
}

// ---------------------------------------------------------------------------
// Attention pass B: res[b, nh*8+e, pix] (f32, BCHW) per (b, head, pix).
// ---------------------------------------------------------------------------
__global__ __launch_bounds__(256) void attn_apply_kernel(
    const bf16* __restrict__ qkvo, const float* __restrict__ stats,
    float* __restrict__ res)
{
  int gid = blockIdx.x * 256 + threadIdx.x;
  int pix = gid & (HW_PIX - 1);
  int bn  = gid >> 14;
  int nh  = bn & 15;
  int b   = bn >> 4;
  const float* st = stats + (size_t)bn * 80;   // uniform per block
  size_t m = (size_t)b * HW_PIX + pix;
  bf16x8 qraw = *(const bf16x8*)(qkvo + (m << 9) + nh * 8);
  float q[8];
#pragma unroll
  for (int d = 0; d < 8; ++d) {
    float v = (float)qraw[d];
    q[d] = v > 0.f ? v + 1.f : __expf(v);
  }
  const float scale = 0.35355339059327373f;     // 8^-0.5
  const float invL  = 1.f / (float)HW_PIX;
  float z = 0.f;
#pragma unroll
  for (int d = 0; d < 8; ++d) z += q[d] * st[64 + d];
  z *= scale * invL;                            // q . mean(k) * scale
  int h = pix >> 7, wq = pix & 127;
  float fhh = (float)h, fww = (float)wq;
  float sn[8], cs[8];
  sn[0] = sn[1] = __sinf(fhh);         cs[0] = cs[1] = __cosf(fhh);
  sn[2] = sn[3] = __sinf(fhh * 1e-4f); cs[2] = cs[3] = __cosf(fhh * 1e-4f);
  sn[4] = sn[5] = __sinf(fww);         cs[4] = cs[5] = __cosf(fww);
  sn[6] = sn[7] = __sinf(fww * 1e-4f); cs[6] = cs[7] = __cosf(fww * 1e-4f);
  float qr[8];
#pragma unroll
  for (int pr = 0; pr < 4; ++pr) {
    float x0 = q[2 * pr], x1 = q[2 * pr + 1];
    qr[2 * pr]     = x0 * cs[2 * pr]     - x1 * sn[2 * pr];
    qr[2 * pr + 1] = x1 * cs[2 * pr + 1] + x0 * sn[2 * pr + 1];
  }
  float f = 1.f + 1.f / (z + 1e-6f);
  const float kvs = scale * invL;
  float* rp = res + (((size_t)b * CDIM + nh * 8) << 14) + pix;
#pragma unroll
  for (int e = 0; e < 8; ++e) {
    float dot = 0.f;
#pragma unroll
    for (int d = 0; d < 8; ++d) dot += qr[d] * st[d * 8 + e];
    rp[(size_t)e << 14] = dot * kvs * f - z * st[72 + e] * invL;
  }
}

// ---------------------------------------------------------------------------
// LDS-tiled 5x5 depthwise (lepe) on v + bias, add res, multiply by o gate,
// emit proj input bf16 [M, 128].
// ---------------------------------------------------------------------------
__global__ __launch_bounds__(256) void lepe_o_kernel(
    const bf16* __restrict__ qkvo, const float* __restrict__ res,
    const float* __restrict__ lw, const float* __restrict__ lb,
    bf16* __restrict__ projin)
{
  __shared__ bf16 patch[20 * 20];
  const int c  = blockIdx.y;
  const int b  = blockIdx.z;
  const int th = (blockIdx.x >> 3) << 4;
  const int tw = (blockIdx.x & 7) << 4;
  const int tid = threadIdx.x;
  for (int i = tid; i < 400; i += 256) {
    int py = i / 20, px = i % 20;
    int hh = th + py - 2, ww = tw + px - 2;
    bf16 v = (bf16)0.f;
    if (hh >= 0 && hh < 128 && ww >= 0 && ww < 128)
      v = qkvo[((size_t)(b * HW_PIX + hh * 128 + ww) << 9) + 256 + c];
    patch[i] = v;
  }
  __syncthreads();
  float wreg[25];
#pragma unroll
  for (int i = 0; i < 25; ++i) wreg[i] = lw[c * 25 + i];
  int ty = tid >> 4, tx = tid & 15;
  float acc = lb[c];
#pragma unroll
  for (int dy = 0; dy < 5; ++dy)
#pragma unroll
    for (int dx = 0; dx < 5; ++dx)
      acc += (float)patch[(ty + dy) * 20 + tx + dx] * wreg[dy * 5 + dx];
  int pix  = (th + ty) * 128 + tw + tx;
  size_t m = (size_t)b * HW_PIX + pix;
  float r  = res[(((size_t)b * CDIM + c) << 14) + pix];
  float o  = (float)qkvo[(m << 9) + 384 + c];
  projin[m * CDIM + c] = (bf16)((r + acc) * o);
}

// ---------------------------------------------------------------------------
// FFN depthwise 3x3 on both halves of p [M, 680] + exact-GELU gate ->
// g_bf [M, 352] (K-padded with zeros for the ffn_out WMMA GEMM).
// ---------------------------------------------------------------------------
__global__ __launch_bounds__(256) void dw_gelu_kernel(
    const bf16* __restrict__ p, const float* __restrict__ dw,
    bf16* __restrict__ g)
{
  int gid = blockIdx.x * 256 + threadIdx.x;
  int pix = gid & (HW_PIX - 1);
  int bj  = gid >> 14;
  int j   = bj % HIDDEN;
  int b   = bj / HIDDEN;
  int h = pix >> 7, wq = pix & 127;
  float a1 = 0.f, a2 = 0.f;
#pragma unroll
  for (int dy = 0; dy < 3; ++dy) {
    int hh = h + dy - 1;
    if (hh < 0 || hh > 127) continue;
#pragma unroll
    for (int dx = 0; dx < 3; ++dx) {
      int ww = wq + dx - 1;
      if (ww < 0 || ww > 127) continue;
      size_t base = ((size_t)b * HW_PIX + hh * 128 + ww) * (size_t)(2 * HIDDEN);
      a1 += (float)p[base + j]          * dw[j * 9 + dy * 3 + dx];
      a2 += (float)p[base + j + HIDDEN] * dw[(j + HIDDEN) * 9 + dy * 3 + dx];
    }
  }
  float ge = 0.5f * a1 * (1.f + erff(a1 * 0.70710678118f));
  size_t m = (size_t)b * HW_PIX + pix;
  g[m * HIDP + j] = (bf16)(ge * a2);
  if (j < HIDP - HIDDEN) g[m * HIDP + HIDDEN + j] = (bf16)0.f;
}

// ---------------------------------------------------------------------------
// Weight conversion helpers.
// ---------------------------------------------------------------------------
__global__ void f32_to_bf16(const float* __restrict__ s, bf16* __restrict__ d, int n) {
  int gid = blockIdx.x * 256 + threadIdx.x;
  if (gid < n) d[gid] = (bf16)s[gid];
}
__global__ void convert_pad(const float* __restrict__ s, bf16* __restrict__ d,
                            int rows, int cols, int ldd) {
  int gid = blockIdx.x * 256 + threadIdx.x;
  if (gid >= rows * ldd) return;
  int r = gid / ldd, c = gid % ldd;
  d[gid] = (c < cols) ? (bf16)s[r * cols + c] : (bf16)0.f;
}

// ---------------------------------------------------------------------------
// Launcher.
// ---------------------------------------------------------------------------
extern "C" void kernel_launch(void* const* d_in, const int* in_sizes, int n_in,
                              void* d_out, int out_size, void* d_ws, size_t ws_size,
                              hipStream_t stream)
{
  (void)in_sizes; (void)n_in; (void)out_size; (void)ws_size;
  const float* x         = (const float*)d_in[0];
  const float* ln1_w     = (const float*)d_in[1];
  const float* ln1_b     = (const float*)d_in[2];
  const float* qkvo_w    = (const float*)d_in[3];
  const float* qkvo_b    = (const float*)d_in[4];
  const float* lepe_w    = (const float*)d_in[5];
  const float* lepe_b    = (const float*)d_in[6];
  const float* proj_w    = (const float*)d_in[7];
  const float* proj_b    = (const float*)d_in[8];
  const float* ln2_w     = (const float*)d_in[9];
  const float* ln2_b     = (const float*)d_in[10];
  const float* ffn_in_w  = (const float*)d_in[11];
  const float* ffn_dw_w  = (const float*)d_in[12];
  const float* ffn_out_w = (const float*)d_in[13];

  uint8_t* ws = (uint8_t*)d_ws;
  const size_t MB = 1ull << 20;
  bf16*  qkvo_bf = (bf16*) (ws + 0);            // [M,512] bf16 (128MB)
  bf16*  p_bf    = (bf16*) (ws + 0);            // [M,680] bf16 (170MB, reuses dead qkvo)
  float* res_f   = (float*)(ws + 192 * MB);     // [B,128,HW] f32 (64MB)
  bf16*  projin  = (bf16*) (ws + 256 * MB);     // [M,128] bf16 (32MB)
  float* x1_f    = (float*)(ws + 288 * MB);     // [B,128,HW] f32 (64MB)
  bf16*  xn_bf   = (bf16*) (ws + 352 * MB);     // [M,128] bf16 (32MB, ln1 then ln2)
  bf16*  g_bf    = (bf16*) (ws + 384 * MB);     // [M,352] bf16 (88MB)
  bf16*  wqkvo   = (bf16*) (ws + 472 * MB);
  bf16*  wproj   = (bf16*) (ws + 472 * MB + 256 * 1024);
  bf16*  wffni   = (bf16*) (ws + 472 * MB + 512 * 1024);
  bf16*  wffno   = (bf16*) (ws + 472 * MB + 1024 * 1024);
  float* stats   = (float*)(ws + 472 * MB + 1536 * 1024);

  dim3 b256(256);

  // weights -> bf16 (ffn_out K-padded 340 -> 352)
  f32_to_bf16<<<dim3(65536 / 256), b256, 0, stream>>>(qkvo_w, wqkvo, 512 * 128);
  f32_to_bf16<<<dim3(16384 / 256), b256, 0, stream>>>(proj_w, wproj, 128 * 128);
  f32_to_bf16<<<dim3((87040 + 255) / 256), b256, 0, stream>>>(ffn_in_w, wffni, 680 * 128);
  convert_pad<<<dim3((128 * HIDP) / 256), b256, 0, stream>>>(ffn_out_w, wffno, 128, HIDDEN, HIDP);

  // LN1 -> xn (bf16)
  ln_kernel<<<dim3(MTOK / 256), b256, 0, stream>>>(x, ln1_w, ln1_b, xn_bf);

  // qkvo GEMM: [M,128] x [512,128]^T + bias -> [M,512] bf16
  gemm_wmma<0><<<dim3(4, MTOK / 128), b256, 0, stream>>>(
      xn_bf, wqkvo, qkvo_b, nullptr, qkvo_bf, nullptr, MTOK, 512, 128, 512);

  // linear attention
  attn_reduce_kernel<<<dim3(NBATCH * 16), b256, 0, stream>>>(qkvo_bf, stats);
  attn_apply_kernel<<<dim3(NBATCH * 16 * HW_PIX / 256), b256, 0, stream>>>(
      qkvo_bf, stats, res_f);

  // lepe 5x5 dw + bias, add, gate by o -> proj input bf16
  lepe_o_kernel<<<dim3(64, 128, NBATCH), b256, 0, stream>>>(
      qkvo_bf, res_f, lepe_w, lepe_b, projin);

  // proj GEMM + bias + residual(x) -> x1 (f32 BCHW)
  gemm_wmma<1><<<dim3(1, MTOK / 128), b256, 0, stream>>>(
      projin, wproj, proj_b, x, nullptr, x1_f, MTOK, 128, 128, 128);

  // LN2 -> xn (bf16, reuse)
  ln_kernel<<<dim3(MTOK / 256), b256, 0, stream>>>(x1_f, ln2_w, ln2_b, xn_bf);

  // ffn_in GEMM: [M,128] x [680,128]^T -> p [M,680] bf16
  gemm_wmma<0><<<dim3(6, MTOK / 128), b256, 0, stream>>>(
      xn_bf, wffni, nullptr, nullptr, p_bf, nullptr, MTOK, 680, 128, 680);

  // 3x3 dw on both halves + GELU gate -> g [M,352] bf16 (zero-padded K)
  dw_gelu_kernel<<<dim3(NBATCH * HIDDEN * HW_PIX / 256), b256, 0, stream>>>(
      p_bf, ffn_dw_w, g_bf);

  // ffn_out GEMM + residual(x1) -> d_out (f32 BCHW)
  gemm_wmma<1><<<dim3(1, MTOK / 128), b256, 0, stream>>>(
      g_bf, wffno, nullptr, x1_f, nullptr, (float*)d_out, MTOK, 128, HIDP, 128);
}
